// CausalSelfAttention_360777253634
// MI455X (gfx1250) — matrix operand order repaired and checked by reading, not compiled, not run on hardware
//
#include <hip/hip_runtime.h>
#include <math.h>

// ---------------- problem constants ----------------
#define HQ   16
#define HKV  4
#define DH   128
#define WINL 1024
#define CDIM 2048

// ---------------- CDNA5 WMMA types ----------------
typedef __attribute__((ext_vector_type(16))) __bf16 v16bf;
typedef __attribute__((ext_vector_type(8)))  float  v8f;
typedef __attribute__((ext_vector_type(4)))  int    v4i;

struct U128 { unsigned int x, y, z, w; };

union BF16x16 {
    v16bf          v;
    U128           u[2];
    unsigned short s[16];
};

static __device__ __forceinline__ unsigned short f2bf(float f) {
    unsigned int u = __builtin_bit_cast(unsigned int, f);
    u += 0x7FFFu + ((u >> 16) & 1u);  // round-to-nearest-even
    return (unsigned short)(u >> 16);
}

static __device__ __forceinline__ v8f v8zero() {
    v8f z;
    for (int i = 0; i < 8; ++i) z[i] = 0.0f;
    return z;
}

static __device__ __forceinline__ v8f wmma_bf16(const v16bf& a, const v16bf& b, const v8f& c) {
    return __builtin_amdgcn_wmma_f32_16x16x32_bf16(false, a, false, b, (short)0, c, false, false);
}

// ---------------- CDNA5 async global->LDS copy (ASYNCcnt-tracked DMA) ----------------
#if __has_builtin(__builtin_amdgcn_global_load_async_to_lds_b128)
#define HAVE_ASYNC_LDS 1
#else
#define HAVE_ASYNC_LDS 0
#endif

static __device__ __forceinline__ void async_b128(const unsigned short* g, unsigned short* l) {
#if HAVE_ASYNC_LDS
    __builtin_amdgcn_global_load_async_to_lds_b128(
        (__attribute__((address_space(1))) v4i*)g,
        (__attribute__((address_space(3))) v4i*)l, 0, 0);
#else
    *(U128*)l = *(const U128*)g;   // sync fallback: load + ds_store
#endif
}

#if HAVE_ASYNC_LDS
#if __has_builtin(__builtin_amdgcn_s_wait_asynccnt)
#define WAIT_ASYNC(n) __builtin_amdgcn_s_wait_asynccnt(n)
#else
#define WAIT_ASYNC(n) asm volatile("s_wait_asynccnt %0" ::"n"(n))
#endif
#else
#define WAIT_ASYNC(n) ((void)0)
#endif

// =====================================================================
// prep: xbf = bf16(x); x0nbf = bf16(rmsnorm(x0))   (one row of C per block)
// =====================================================================
__global__ __launch_bounds__(256) void prep_kernel(
    const float* __restrict__ x, const float* __restrict__ x0,
    unsigned short* __restrict__ xbf, unsigned short* __restrict__ x0nbf, int C)
{
    __shared__ float red[256];
    size_t row = blockIdx.x;
    const float* xr  = x  + row * C;
    const float* x0r = x0 + row * C;
    float ss = 0.0f;
    for (int c = threadIdx.x; c < C; c += 256) {
        float v = x0r[c];
        ss += v * v;
        xbf[row * C + c] = f2bf(xr[c]);
    }
    red[threadIdx.x] = ss;
    __syncthreads();
    for (int off = 128; off > 0; off >>= 1) {
        if ((int)threadIdx.x < off) red[threadIdx.x] += red[threadIdx.x + off];
        __syncthreads();
    }
    float r = rsqrtf(red[0] / (float)C + 1.1920929e-07f);
    for (int c = threadIdx.x; c < C; c += 256)
        x0nbf[row * C + c] = f2bf(x0r[c] * r);
}

// =====================================================================
// elementwise f32 -> bf16 convert (grid-stride)
// =====================================================================
__global__ __launch_bounds__(256) void cvt_bf16_kernel(
    const float* __restrict__ in, unsigned short* __restrict__ out, size_t n)
{
    size_t i = (size_t)blockIdx.x * blockDim.x + threadIdx.x;
    size_t st = (size_t)gridDim.x * blockDim.x;
    for (; i < n; i += st) out[i] = f2bf(in[i]);
}

// =====================================================================
// bf16 WMMA GEMM:  C[M,N] = A[M,K] * B[N,K]^T    (fp32 accumulate/out)
// 128x128 block tile, 8 waves (4 row-groups x 2 col-halves, 32x64 per wave);
// async double-buffered LDS staging, K-step 32.
// =====================================================================
#define GTK        32
#define LDS_STRIDE 40   // 32 + 8 pad (keeps 16B alignment of 8-half chunks)

__global__ __launch_bounds__(256) void gemm_bf16_kernel(
    const unsigned short* __restrict__ A,
    const unsigned short* __restrict__ Bm,
    float* __restrict__ C, int M, int N, int K, int ldc)
{
    __shared__ unsigned short As[2][128 * LDS_STRIDE];
    __shared__ unsigned short Bs[2][128 * LDS_STRIDE];

    const int tid  = threadIdx.x;
    const int wave = tid >> 5;
    const int wr   = wave >> 1;      // row group: 32 rows
    const int wc   = wave & 1;       // col half:  64 cols
    const int lane = tid & 31;
    const int half = lane >> 4;
    const int l16  = lane & 15;
    const int rowBase = blockIdx.y * 128;
    const int colBase = blockIdx.x * 128;

    v8f acc[2][4];
    for (int mi = 0; mi < 2; ++mi)
        for (int nt = 0; nt < 4; ++nt) acc[mi][nt] = v8zero();

    // 4 async b128 per thread per fill (A chunk x2 + B chunk x2)
    auto issue_fill = [&](int buf, int k0) {
        for (int it2 = 0; it2 < 2; ++it2) {
            int linear = tid + it2 * 256;        // 0..511 chunks of 8 halfs
            int r      = linear >> 2;            // 4 chunks / row
            int ch     = (linear & 3) * 8;
            async_b128(A  + (size_t)(rowBase + r) * K + k0 + ch, &As[buf][r * LDS_STRIDE + ch]);
            async_b128(Bm + (size_t)(colBase + r) * K + k0 + ch, &Bs[buf][r * LDS_STRIDE + ch]);
        }
    };

    const int nIter = K / GTK;
    issue_fill(0, 0);

    for (int it = 0; it < nIter; ++it) {
        const int cur = it & 1;
        if (it + 1 < nIter) {
            issue_fill(1 - cur, (it + 1) * GTK);
            WAIT_ASYNC(4);                 // current buffer's 4 copies done
        } else {
            WAIT_ASYNC(0);
        }
        __syncthreads();                   // whole block's fill of `cur` done

        // two A fragments (rows wr*32 + {0,16} + l16)
        BF16x16 af[2];
        for (int mi = 0; mi < 2; ++mi) {
            const int ar = wr * 32 + mi * 16 + l16;
            af[mi].u[0] = *(const U128*)(&As[cur][ar * LDS_STRIDE + half * 8]);
            af[mi].u[1] = *(const U128*)(&As[cur][ar * LDS_STRIDE + 16 + half * 8]);
        }
        // four B fragments (cols wc*64 + nt*16 + l16)
        BF16x16 bf[4];
        for (int nt = 0; nt < 4; ++nt) {
            const int br = wc * 64 + nt * 16 + l16;
            bf[nt].u[0] = *(const U128*)(&Bs[cur][br * LDS_STRIDE + half * 8]);
            bf[nt].u[1] = *(const U128*)(&Bs[cur][br * LDS_STRIDE + 16 + half * 8]);
        }
        // 8 WMMAs with 2x4 fragment reuse
        for (int mi = 0; mi < 2; ++mi)
            for (int nt = 0; nt < 4; ++nt)
                acc[mi][nt] = wmma_bf16(af[mi].v, bf[nt].v, acc[mi][nt]);

        __syncthreads();                   // all waves done with `cur` before overwrite
    }

    // epilogue: D layout = lane holds col l16, VGPR v holds row v+8*half
    for (int mi = 0; mi < 2; ++mi) {
        for (int nt = 0; nt < 4; ++nt) {
            int c = colBase + wc * 64 + nt * 16 + l16;
            for (int v = 0; v < 8; ++v) {
                int r = rowBase + wr * 32 + mi * 16 + v + 8 * half;
                if (r < M && c < N) C[(size_t)r * ldc + c] = acc[mi][nt][v];
            }
        }
    }
}

// =====================================================================
// combine: q/k/v = alpha*P + beta*P0; RoPE + rmsnorm*1.2 for q,k; v_init out
// grid (24, T, B): unit 0..15 = q head, 16..19 = k head, 20..23 = v head
// =====================================================================
__global__ __launch_bounds__(128) void combine_kernel(
    const float* __restrict__ P, const float* __restrict__ P0,
    const float* __restrict__ cosb, const float* __restrict__ sinb,
    const float* aq, const float* bq, const float* ak, const float* bk,
    const float* av, const float* bv,
    unsigned short* __restrict__ qout, unsigned short* __restrict__ kout,
    unsigned short* __restrict__ vout, float* __restrict__ vinit, int T)
{
    __shared__ float buf[DH];
    __shared__ float red[DH];

    const int unit = blockIdx.x;
    const int t = blockIdx.y, b = blockIdx.z;
    const int d = threadIdx.x;
    const size_t rowoff = ((size_t)b * T + t) * 3072;

    if (unit >= 20) {  // ---- v heads ----
        int kh  = unit - 20;
        int col = 2560 + kh * DH + d;
        float pv  = P[rowoff + col];
        float val = av[0] * pv + bv[0] * P0[rowoff + col];
        size_t oi = (((size_t)b * T + t) * HKV + kh) * DH + d;
        vinit[oi] = pv;                 // v before mixing (second output)
        vout[oi]  = f2bf(val);
        return;
    }

    bool isq = unit < 16;
    int col; float Am, Bc; size_t oi; unsigned short* outp;
    if (isq) {
        col = unit * DH + d; Am = aq[0]; Bc = bq[0];
        oi = (((size_t)b * T + t) * HQ + unit) * DH + d; outp = qout;
    } else {
        int kh = unit - 16;
        col = 2048 + kh * DH + d; Am = ak[0]; Bc = bk[0];
        oi = (((size_t)b * T + t) * HKV + kh) * DH + d; outp = kout;
    }
    buf[d] = Am * P[rowoff + col] + Bc * P0[rowoff + col];
    __syncthreads();

    float rot;
    if (d < 64) {
        float c = cosb[t * 64 + d], s = sinb[t * 64 + d];
        rot = buf[d] * c + buf[d + 64] * s;
    } else {
        float c = cosb[t * 64 + d - 64], s = sinb[t * 64 + d - 64];
        rot = -buf[d - 64] * s + buf[d] * c;
    }
    red[d] = rot * rot;
    __syncthreads();
    for (int off = 64; off > 0; off >>= 1) {
        if (d < off) red[d] += red[d + off];
        __syncthreads();
    }
    float r = rsqrtf(red[0] / (float)DH + 1.1920929e-07f);
    outp[oi] = f2bf(rot * r * 1.2f);
}

// =====================================================================
// flash attention (sliding-window causal GQA), one wave per 16-row q tile
// V tiles staged global->LDS asynchronously (double buffered per wave)
// =====================================================================
#define AT_WAVES 4
#define VSTRIDE  136    // 128 + 8 halfs pad; 272B row = 17x16B (keeps alignment)

__global__ __launch_bounds__(AT_WAVES * 32) void attn_kernel(
    const unsigned short* __restrict__ qbf,   // (B,T,HQ,DH) bf16
    const unsigned short* __restrict__ kbf,   // (B,T,HKV,DH) bf16
    const unsigned short* __restrict__ vbf,   // (B,T,HKV,DH) bf16
    unsigned short* __restrict__ ybf,         // (B,T,HQ,DH) bf16
    int T)
{
    __shared__ unsigned short Ps[AT_WAVES][16 * LDS_STRIDE];
    __shared__ unsigned short Vs[AT_WAVES][2][32 * VSTRIDE];

    const int wave = threadIdx.x >> 5;
    const int lane = threadIdx.x & 31;
    const int half = lane >> 4;
    const int l16  = lane & 15;
    const int qt   = blockIdx.x * AT_WAVES + wave;   // wave-uniform
    const int head = blockIdx.y;
    const int b    = blockIdx.z;
    const int ntiles = T / 16;
    if (qt >= ntiles) return;                        // wave-uniform exit
    const int qbase = qt * 16;
    const int hk    = head >> 2;                     // HQ/HKV = 4
    const float scale = 0.08838834764831845f;        // 1/sqrt(128)
    const float NEG_INF = -__builtin_inff();

    // Q fragments: A-layout, M = l16, K-halves split by lane half
    BF16x16 aq[4];
    {
        const unsigned short* qrow =
            qbf + (((size_t)b * T + qbase + l16) * HQ + head) * DH;
        for (int kk = 0; kk < 4; ++kk) {
            aq[kk].u[0] = *(const U128*)(qrow + kk * 32 + half * 16);
            aq[kk].u[1] = *(const U128*)(qrow + kk * 32 + half * 16 + 8);
        }
    }

    v8f O[8];
    for (int i = 0; i < 8; ++i) O[i] = v8zero();
    float m[8], l[8];
    for (int v = 0; v < 8; ++v) { m[v] = NEG_INF; l[v] = 0.0f; }

    int jmin = qbase - WINL; if (jmin < 0) jmin = 0;
    const int kb0  = jmin & ~31;
    const int kend = qbase + 16;

    // async-stage a 32x64 V slice (d 0..63) (one kv row per lane, 8 x 16B)
    auto issue_v = [&](int buf, int kb) {
        int kvr = kb + lane; if (kvr > T - 1) kvr = T - 1;
        const unsigned short* vrow = vbf + (((size_t)b * T + kvr) * HKV + hk) * DH;
        unsigned short* dst = &Vs[wave][buf][lane * VSTRIDE];
        for (int c = 0; c < 8; ++c) async_b128(vrow + c * 8, dst + c * 8);
    };

    issue_v(0, kb0);

    for (int kb = kb0; kb < kend; kb += 32) {
        const int cur = ((kb - kb0) >> 5) & 1;

        // ---- S = Q * K^T over this 32-key block (two 16x16 tiles) ----
        v8f S0 = v8zero(), S1 = v8zero();
        for (int kk = 0; kk < 4; ++kk) {
            int kv0 = kb + l16;       if (kv0 > T - 1) kv0 = T - 1;
            int kv1 = kb + 16 + l16;  if (kv1 > T - 1) kv1 = T - 1;
            const unsigned short* kr0 =
                kbf + (((size_t)b * T + kv0) * HKV + hk) * DH + kk * 32 + half * 16;
            const unsigned short* kr1 =
                kbf + (((size_t)b * T + kv1) * HKV + hk) * DH + kk * 32 + half * 16;
            BF16x16 b0, b1;
            b0.u[0] = *(const U128*)(kr0); b0.u[1] = *(const U128*)(kr0 + 8);
            b1.u[0] = *(const U128*)(kr1); b1.u[1] = *(const U128*)(kr1 + 8);
            S0 = wmma_bf16(aq[kk].v, b0.v, S0);
            S1 = wmma_bf16(aq[kk].v, b1.v, S1);
        }

        // ---- mask + online softmax (row stats within 16-lane half) ----
        float corr[8], p0[8], p1[8];
        for (int v = 0; v < 8; ++v) {
            int i  = qbase + v + 8 * half;
            int j0 = kb + l16;
            int j1 = kb + 16 + l16;
            float s0 = (j0 <= i && i - j0 <= WINL) ? S0[v] * scale : NEG_INF;
            float s1 = (j1 <= i && i - j1 <= WINL) ? S1[v] * scale : NEG_INF;
            float mt = fmaxf(s0, s1);
            for (int off = 1; off < 16; off <<= 1)
                mt = fmaxf(mt, __shfl_xor(mt, off, 32));
            float mn = fmaxf(m[v], mt);
            float c  = (m[v] > NEG_INF) ? __expf(m[v] - mn) : 0.0f;
            p0[v] = (s0 > NEG_INF) ? __expf(s0 - mn) : 0.0f;
            p1[v] = (s1 > NEG_INF) ? __expf(s1 - mn) : 0.0f;
            float rs = p0[v] + p1[v];
            for (int off = 1; off < 16; off <<= 1)
                rs += __shfl_xor(rs, off, 32);
            l[v] = l[v] * c + rs;
            m[v] = mn;
            corr[v] = c;
        }
        for (int nt = 0; nt < 8; ++nt)
            for (int v = 0; v < 8; ++v) O[nt][v] *= corr[v];

        // ---- re-layout P (D-layout -> A-layout) via per-wave LDS ----
        unsigned short* pp = Ps[wave];
        for (int v = 0; v < 8; ++v) {
            int r = v + 8 * half;
            pp[r * LDS_STRIDE + l16]      = f2bf(p0[v]);
            pp[r * LDS_STRIDE + 16 + l16] = f2bf(p1[v]);
        }
        BF16x16 ap;
        ap.u[0] = *(const U128*)(pp + l16 * LDS_STRIDE + half * 16);
        ap.u[1] = *(const U128*)(pp + l16 * LDS_STRIDE + half * 16 + 8);

        // ---- software-pipeline the next V tile, wait for current ----
        if (kb + 32 < kend) {
            issue_v(1 - cur, kb + 32);
            WAIT_ASYNC(8);                 // current tile's 8 copies complete
        } else {
            WAIT_ASYNC(0);
        }

        // ---- O += P * V  (B-frag gathered from LDS: col d, K = kv) ----
        const unsigned short* vs = Vs[wave][cur];
        for (int nt = 0; nt < 8; ++nt) {
            BF16x16 bv;
            for (int e = 0; e < 16; ++e)
                bv.s[e] = vs[(half * 16 + e) * VSTRIDE + nt * 16 + l16];
            O[nt] = wmma_bf16(ap.v, bv.v, O[nt]);
        }
    }

    // ---- normalize + store bf16 ----
    for (int nt = 0; nt < 8; ++nt) {
        for (int v = 0; v < 8; ++v) {
            int r = qbase + v + 8 * half;
            float o = O[nt][v] / l[v];
            ybf[(((size_t)b * T + r) * HQ + head) * DH + nt * 16 + l16] = f2bf(o);
        }
    }
}

// =====================================================================
// host launcher
// =====================================================================
extern "C" void kernel_launch(void* const* d_in, const int* in_sizes, int n_in,
                              void* d_out, int out_size, void* d_ws, size_t ws_size,
                              hipStream_t stream)
{
    const float* x    = (const float*)d_in[0];
    const float* x0   = (const float*)d_in[1];
    // d_in[2] = ve (all zeros, unused by reference math path)
    const float* cosb = (const float*)d_in[3];
    const float* sinb = (const float*)d_in[4];
    const float* Wq   = (const float*)d_in[5];
    const float* Wk   = (const float*)d_in[6];
    const float* Wv   = (const float*)d_in[7];
    const float* Wp   = (const float*)d_in[8];
    const float* aq   = (const float*)d_in[9];
    const float* bq   = (const float*)d_in[10];
    const float* ak   = (const float*)d_in[11];
    const float* bk   = (const float*)d_in[12];
    const float* av   = (const float*)d_in[13];
    const float* bv   = (const float*)d_in[14];

    const int C = CDIM;
    const int T = in_sizes[3] / (DH / 2);            // cos is (T, 64)
    const int B = in_sizes[0] / (T * C);
    const int M = B * T;                              // 8192
    const int NQKV = HQ * DH + 2 * HKV * DH;          // 3072

    // ---- workspace layout (256B aligned) ----
    char* ws = (char*)d_ws;
    size_t off = 0;
    auto take = [&](size_t bytes) -> char* {
        char* p = ws + off;
        off = (off + bytes + 255) & ~(size_t)255;
        return p;
    };
    unsigned short* xbf    = (unsigned short*)take((size_t)M * C * 2);
    unsigned short* x0nbf  = (unsigned short*)take((size_t)M * C * 2);
    unsigned short* wqkv   = (unsigned short*)take((size_t)NQKV * C * 2);
    unsigned short* wproj  = (unsigned short*)take((size_t)C * C * 2);
    float*          P      = (float*)take((size_t)M * NQKV * 4);
    float*          P0     = (float*)take((size_t)M * NQKV * 4);
    unsigned short* qbuf   = (unsigned short*)take((size_t)M * HQ  * DH * 2);
    unsigned short* kbuf   = (unsigned short*)take((size_t)M * HKV * DH * 2);
    unsigned short* vbuf   = (unsigned short*)take((size_t)M * HKV * DH * 2);
    unsigned short* ybuf   = (unsigned short*)take((size_t)M * HQ  * DH * 2);

    float* yout  = (float*)d_out;                     // (B,T,C)
    float* vinit = yout + (size_t)M * C;              // (B,T,HKV,DH)

    // 1) x -> bf16 ; rmsnorm(x0) -> bf16
    prep_kernel<<<M, 256, 0, stream>>>(x, x0, xbf, x0nbf, C);

    // 2) weights -> bf16 (Wq|Wk|Wv packed along N)
    cvt_bf16_kernel<<<2048, 256, 0, stream>>>(Wq, wqkv,                     (size_t)HQ  * DH * C);
    cvt_bf16_kernel<<<512,  256, 0, stream>>>(Wk, wqkv + (size_t)2048 * C,  (size_t)HKV * DH * C);
    cvt_bf16_kernel<<<512,  256, 0, stream>>>(Wv, wqkv + (size_t)2560 * C,  (size_t)HKV * DH * C);
    cvt_bf16_kernel<<<2048, 256, 0, stream>>>(Wp, wproj,                    (size_t)C * C);

    // 3) P = x @ [Wq;Wk;Wv]^T ; P0 = rmsnorm(x0) @ [Wq;Wk;Wv]^T
    dim3 gg(NQKV / 128, M / 128);
    gemm_bf16_kernel<<<gg, 256, 0, stream>>>(xbf,   wqkv, P,  M, NQKV, C, NQKV);
    gemm_bf16_kernel<<<gg, 256, 0, stream>>>(x0nbf, wqkv, P0, M, NQKV, C, NQKV);

    // 4) mix + RoPE + rmsnorm -> bf16 q/k/v ; write v_init
    combine_kernel<<<dim3(24, T, B), 128, 0, stream>>>(
        P, P0, cosb, sinb, aq, bq, ak, bk, av, bv,
        qbuf, kbuf, vbuf, vinit, T);

    // 5) sliding-window flash attention
    attn_kernel<<<dim3(T / 16 / AT_WAVES, HQ, B), AT_WAVES * 32, 0, stream>>>(
        qbuf, kbuf, vbuf, ybuf, T);

    // 6) y = attn_out @ Wproj^T  (fp32 out)
    gemm_bf16_kernel<<<dim3(C / 128, M / 128), 256, 0, stream>>>(
        ybuf, wproj, yout, M, C, C, C);
}
